// RNN_88210038325690
// MI455X (gfx1250) — matrix operand order repaired and checked by reading, not compiled, not run on hardware
//
#include <hip/hip_runtime.h>

#define SEQ   8192
#define EMB   64
#define NLEAF 8192

typedef __attribute__((ext_vector_type(16))) __bf16 v16bf;
typedef __attribute__((ext_vector_type(8)))  __bf16 v8bf;
typedef __attribute__((ext_vector_type(8)))  float  v8f;
typedef __attribute__((ext_vector_type(4)))  float  v4f;

// Convert 4 fp32 -> bf16 hi/lo split fragments (bf16x3 precision scheme).
static __device__ __forceinline__ void cvt4(const v4f v, v16bf& hi, v16bf& lo, const int base) {
#pragma unroll
    for (int i = 0; i < 4; ++i) {
        float f  = v[i];
        __bf16 h = (__bf16)f;
        hi[base + i] = h;
        lo[base + i] = (__bf16)(f - (float)h);
    }
}

static __device__ __forceinline__ v16bf load16bf(const __bf16* __restrict__ p) {
    v8bf t0 = *(const v8bf*)(p);
    v8bf t1 = *(const v8bf*)(p + 16);
    return __builtin_shufflevector(t0, t1, 0, 1, 2, 3, 4, 5, 6, 7,
                                           8, 9, 10, 11, 12, 13, 14, 15);
}

// ---------------------------------------------------------------------------
// Kernel 0: one-time emb_W fp32 -> bf16 hi/lo planes (L2-resident, reused by
// every wave of the leaf GEMM -> removes all B-side conversion VALU).
// ---------------------------------------------------------------------------
__global__ __launch_bounds__(256) void convert_W(
    const float* __restrict__ W, __bf16* __restrict__ Whi,
    __bf16* __restrict__ Wlo)
{
    const int i = blockIdx.x * 256 + threadIdx.x;   // EMB*SEQ total
    const float f = W[i];
    const __bf16 h = (__bf16)f;
    Whi[i] = h;
    Wlo[i] = (__bf16)(f - (float)h);
}

// ---------------------------------------------------------------------------
// Kernel 1: leaf embedding GEMM  h0 = leaf[8192,8192] x embW[64,8192]^T
// bf16x3 WMMA (Ah*Bh + Ah*Bl + Al*Bh), f32 accumulate, K split across
// gridDim.y into two partial buffers (deterministic; no float atomics).
// Per wave: 16 rows x 64 cols, K stepped by 32. B loaded pre-converted.
// ---------------------------------------------------------------------------
__global__ __launch_bounds__(256) void leaf_gemm(
    const float* __restrict__ leaf,     // [NLEAF, SEQ] fp32 (streamed, 256MB)
    const __bf16* __restrict__ Whi,     // [EMB, SEQ] bf16 hi
    const __bf16* __restrict__ Wlo,     // [EMB, SEQ] bf16 lo
    float* __restrict__ part)           // [2, NLEAF, EMB] partial sums
{
    const int lane  = threadIdx.x & 31;
    const int wave  = threadIdx.x >> 5;
    const int half  = lane >> 4;        // 0: lanes 0-15, 1: lanes 16-31
    const int l16   = lane & 15;
    const int m0    = (blockIdx.x * 8 + wave) * 16;
    const int khalf = blockIdx.y;

    const float* __restrict__ arow = leaf + (size_t)(m0 + l16) * SEQ;

    v8f acc[4] = {};

    const int ksbeg = khalf * (SEQ / 2);
    const int ksend = ksbeg + (SEQ / 2);

#pragma unroll 1
    for (int ks = ksbeg; ks < ksend; ks += 32) {
        // A fragment 16x32 bf16: lane<16 holds K {0..7,16..23}, lane>=16 {8..15,24..31}
        const int ka = ks + half * 8;
        v16bf ah, al;
        cvt4(*(const v4f*)(arow + ka),      ah, al, 0);
        cvt4(*(const v4f*)(arow + ka + 4),  ah, al, 4);
        cvt4(*(const v4f*)(arow + ka + 16), ah, al, 8);
        cvt4(*(const v4f*)(arow + ka + 20), ah, al, 12);
        __builtin_prefetch(arow + ka + 256, 0, 1);

        // B fragment 32x16 bf16: lane holds column e, K {0..7,16..23} / {8..15,24..31} (pre-split)
        const size_t kb = (size_t)l16 * SEQ + (ks + half * 8);
#pragma unroll
        for (int nt = 0; nt < 4; ++nt) {
            const size_t boff = (size_t)nt * 16 * SEQ + kb;
            const v16bf bh = load16bf(Whi + boff);
            const v16bf bl = load16bf(Wlo + boff);

            acc[nt] = __builtin_amdgcn_wmma_f32_16x16x32_bf16(
                false, ah, false, bh, (short)0, acc[nt], false, false);
            acc[nt] = __builtin_amdgcn_wmma_f32_16x16x32_bf16(
                false, ah, false, bl, (short)0, acc[nt], false, false);
            acc[nt] = __builtin_amdgcn_wmma_f32_16x16x32_bf16(
                false, al, false, bh, (short)0, acc[nt], false, false);
        }
    }

    // C/D layout: VGPR r -> row m0 + r + 8*half, lane%16 -> col
    float* __restrict__ pout = part + (size_t)khalf * NLEAF * EMB;
#pragma unroll
    for (int nt = 0; nt < 4; ++nt) {
        const int e = nt * 16 + l16;
#pragma unroll
        for (int r = 0; r < 8; ++r) {
            const int mm = m0 + r + half * 8;
            pout[(size_t)mm * EMB + e] = acc[nt][r];
        }
    }
}

// ---------------------------------------------------------------------------
// Kernel 2: combine the two K-half partials, add bias, relu -> h0
// ---------------------------------------------------------------------------
__global__ __launch_bounds__(256) void bias_relu_combine(
    const float* __restrict__ part, const float* __restrict__ bias,
    float* __restrict__ h0)
{
    const size_t i = (size_t)blockIdx.x * 256 + threadIdx.x;
    const float v = part[i] + part[(size_t)NLEAF * EMB + i] + bias[i & (EMB - 1)];
    h0[i] = v > 0.f ? v : 0.f;
}

// ---------------------------------------------------------------------------
// Kernel 3: one tree level (exact fp32): out[n,e] = relu(b[e]
//            + sum_j hin[2n,j]*W[e,j] + hin[2n+1,j]*W[e,64+j])
// W staged in LDS padded to 129 floats/row -> conflict-free banks.
// Block: 256 threads = 4 output rows x 64 cols.
// ---------------------------------------------------------------------------
__global__ __launch_bounds__(256) void tree_level(
    const float* __restrict__ hin, const float* __restrict__ W,
    const float* __restrict__ b, float* __restrict__ hout, int M2)
{
    __shared__ float lW[64][129];
    __shared__ float lh[8][64];
    const int tid = threadIdx.x;

    for (int i = tid; i < 64 * 128; i += 256) lW[i >> 7][i & 127] = W[i];

    const int base_row = blockIdx.x * 8;
    for (int i = tid; i < 8 * 64; i += 256) {
        const int r = i >> 6, c = i & 63;
        const int gr = base_row + r;
        lh[r][c] = (gr < 2 * M2) ? hin[(size_t)gr * 64 + c] : 0.f;
    }
    __syncthreads();

    const int nloc = tid >> 6;      // 0..3
    const int e    = tid & 63;
    const int n    = blockIdx.x * 4 + nloc;
    if (n < M2) {
        float acc = b[e];
        const float* wr = &lW[e][0];
        const float* hl = &lh[2 * nloc][0];
        const float* hr = &lh[2 * nloc + 1][0];
#pragma unroll
        for (int j = 0; j < 64; ++j) acc += hl[j] * wr[j];
#pragma unroll
        for (int j = 0; j < 64; ++j) acc += hr[j] * wr[64 + j];
        hout[(size_t)n * 64 + e] = acc > 0.f ? acc : 0.f;
    }
}

// ---------------------------------------------------------------------------
// Kernel 4: fused tail — levels with 64 input rows down to the root, plus the
// final projection, all in one block with __syncthreads between levels.
// ---------------------------------------------------------------------------
__global__ __launch_bounds__(256) void tree_tail(
    const float* __restrict__ hin,   // [64, 64]
    const float* __restrict__ W, const float* __restrict__ b,
    const float* __restrict__ pW, const float* __restrict__ pb,
    float* __restrict__ out)         // [2]
{
    __shared__ float lW[64][129];
    __shared__ float ha[64][64];
    __shared__ float hb[32][64];
    const int tid = threadIdx.x;

    for (int i = tid; i < 64 * 128; i += 256) lW[i >> 7][i & 127] = W[i];
    for (int i = tid; i < 64 * 64; i += 256)  ha[i >> 6][i & 63] = hin[i];
    __syncthreads();

    float (*src)[64] = ha;
    float (*dst)[64] = hb;
#pragma unroll 1
    for (int M2 = 32; M2 >= 1; M2 >>= 1) {
        const int total = M2 * 64;
        for (int i = tid; i < total; i += 256) {
            const int n = i >> 6, e = i & 63;
            float acc = b[e];
            const float* wr = &lW[e][0];
#pragma unroll
            for (int j = 0; j < 64; ++j) acc += src[2 * n][j] * wr[j];
#pragma unroll
            for (int j = 0; j < 64; ++j) acc += src[2 * n + 1][j] * wr[64 + j];
            dst[n][e] = acc > 0.f ? acc : 0.f;
        }
        __syncthreads();
        float (*t)[64] = src; src = dst; dst = t;
    }

    if (tid < 2) {
        float acc = pb[tid];
#pragma unroll
        for (int j = 0; j < 64; ++j) acc += src[0][j] * pW[tid * 64 + j];
        out[tid] = acc;
    }
}

// ---------------------------------------------------------------------------
extern "C" void kernel_launch(void* const* d_in, const int* in_sizes, int n_in,
                              void* d_out, int out_size, void* d_ws, size_t ws_size,
                              hipStream_t stream) {
    const float* leaf = (const float*)d_in[0];
    const float* embW = (const float*)d_in[1];
    const float* embB = (const float*)d_in[2];
    const float* WW   = (const float*)d_in[3];
    const float* Wb   = (const float*)d_in[4];
    const float* pW   = (const float*)d_in[5];
    const float* pb   = (const float*)d_in[6];
    float* out = (float*)d_out;

    char* ws = (char*)d_ws;
    const size_t H = (size_t)NLEAF * EMB * sizeof(float);   // 2 MB
    float*  part = (float*)ws;                              // 2 x 2 MB partials
    float*  bufA = (float*)(ws + 2 * H);                    // 2 MB
    float*  bufB = (float*)(ws + 3 * H);                    // 1 MB
    __bf16* Whi  = (__bf16*)(ws + 4 * H);                   // 1 MB
    __bf16* Wlo  = (__bf16*)(ws + 4 * H + (size_t)EMB * SEQ * sizeof(__bf16));

    // Prologue: split emb_W into bf16 hi/lo planes (once, L2-resident)
    convert_W<<<(EMB * SEQ) / 256, 256, 0, stream>>>(embW, Whi, Wlo);

    // Leaf GEMM: 64 M-blocks x 2 K-halves = 1024 waves of bf16x3 WMMA
    leaf_gemm<<<dim3(NLEAF / (8 * 16), 2), 256, 0, stream>>>(leaf, Whi, Wlo, part);
    bias_relu_combine<<<(NLEAF * EMB) / 256, 256, 0, stream>>>(part, embB, bufA);

    // Tree levels with >=128 input rows: 7 launches, ping-pong buffers
    const float* cur = bufA;
    float* nxt = bufB;
    for (int M2 = NLEAF / 2; M2 >= 64; M2 >>= 1) {
        tree_level<<<dim3((M2 + 3) / 4), 256, 0, stream>>>(cur, WW, Wb, nxt, M2);
        float* t = (float*)cur; cur = nxt; nxt = t;
    }

    // Fused tail: last 6 levels + projection, single block
    tree_tail<<<1, 256, 0, stream>>>(cur, WW, Wb, pW, pb, out);
}
